// EnhancedTemporalGNN_76836964926210
// MI455X (gfx1250) — matrix operand order repaired and checked by reading, not compiled, not run on hardware
//
#include <hip/hip_runtime.h>
#include <math.h>

// ---------------------------------------------------------------------------
// Types for CDNA5 WMMA
// ---------------------------------------------------------------------------
typedef __attribute__((ext_vector_type(16))) __bf16 bf16x16;
typedef __attribute__((ext_vector_type(8)))  float  floatx8;

#define NN_ 16384
#define EE_ 131072
#define DD_ 256
#define HH_ 8
#define HD_ 64

// ---------------------------------------------------------------------------
// Small device helpers
// ---------------------------------------------------------------------------
__device__ __forceinline__ unsigned short f2bf(float f) {
    unsigned int u = __float_as_uint(f);
    u += 0x7FFFu + ((u >> 16) & 1u);          // round-to-nearest-even
    return (unsigned short)(u >> 16);
}
__device__ __forceinline__ float bf2f(unsigned short s) {
    return __uint_as_float(((unsigned int)s) << 16);
}
__device__ __forceinline__ float gelu_f(float x) {
    return 0.5f * x * (1.0f + erff(x * 0.70710678118654752440f));
}
__device__ __forceinline__ void atomicMaxF(float* addr, float v) {
    if (v >= 0.0f) atomicMax((int*)addr, __float_as_int(v));
    else           atomicMin((unsigned int*)addr, __float_as_uint(v));
}

// ---------------------------------------------------------------------------
// Generic bf16 WMMA GEMM:  C[m,n] = act( A[m,:] @ W + b + optional G[gidx[m],n] )
//
//  * 128x64 block tile, 8 wave32 waves, each wave owns a 16x64 strip:
//    one A fragment feeds FOUR v_wmma_f32_16x16x32_bf16 per K-step,
//    all fragments loaded first (2+8 ds_load_b128) -> single dscnt wait.
//  * Double-buffered LDS: next K-tile staged into registers while the
//    current one feeds the WMMAs; ONE barrier per K-step.
//  * A rows may come from two sources split at `ksplit` (K-concat GEMM),
//    each optionally row-gathered; gather indices hoisted out of the K-loop.
//  * W fp32->bf16 staged as packed dword pairs (ds_store_b32).
// ---------------------------------------------------------------------------
__global__ __launch_bounds__(256) void k_gemm(
    const unsigned short* __restrict__ A0, int lda0,
    const unsigned short* __restrict__ A1, int lda1,
    const int* __restrict__ idx0, const int* __restrict__ idx1,
    int ksplit, int Ktot,
    const float* __restrict__ W0, const float* __restrict__ W1, int ldw,
    const float* __restrict__ b0, const float* __restrict__ b1,
    const float* __restrict__ G, const int* __restrict__ gidx, int ldg,
    int act,                                   // 0 none, 1 leaky(0.2), 2 gelu
    unsigned short* __restrict__ Cb, float* __restrict__ C32, int ldc)
{
    __shared__ __align__(16) unsigned short sA[2][128 * 32];   // [m][k]
    __shared__ __align__(16) unsigned short sW[2][64 * 32];    // [n][k]

    const int tid  = threadIdx.x;
    const int lane = tid & 31;
    const int wm   = tid >> 5;          // wave 0..7 -> 16-row band
    const int bx = blockIdx.x, by = blockIdx.y;

    const floatx8 vzero = {0.f,0.f,0.f,0.f,0.f,0.f,0.f,0.f};
    floatx8 acc[4] = {vzero, vzero, vzero, vzero};

    // ---- A staging map: each thread stages 16 halves of one row ----
    const int arow = tid >> 1;                  // 0..127
    const int acol = (tid & 1) * 16;            // 0 or 16
    const int row  = by * 128 + arow;
    int rr0 = row, rr1 = row;
    if (idx0) rr0 = idx0[row];                  // hoisted: K-invariant
    if (idx1) rr1 = idx1[row];
    const unsigned short* aptrA = A0 + (size_t)rr0 * lda0;
    const unsigned short* aptrB = A1 ? (A1 + (size_t)rr1 * lda1 - ksplit) : aptrA;

    // ---- W staging map: thread owns column nn, K-pairs kk = 2*wp + 8*i ----
    const int nn = tid & 63;
    const int wp = tid >> 6;                    // 0..3

    const int nk = Ktot >> 5;
    uint4 ra0, ra1;
    float wf[8];

    // ---- prologue: stage K-step 0 into registers (k0=0 < ksplit always) ----
    {
        ra0 = *(const uint4*)(aptrA + acol);
        ra1 = *(const uint4*)(aptrA + acol + 8);
        const float* wbase = W0 + bx * 64 + nn;
        #pragma unroll
        for (int i = 0; i < 4; ++i) {
            int kk = 2 * wp + 8 * i;
            wf[2 * i]     = wbase[(size_t)kk * ldw];
            wf[2 * i + 1] = wbase[(size_t)(kk + 1) * ldw];
        }
    }

    int cur = 0;
    for (int ks = 0; ks < nk; ++ks) {
        // ---- commit staged registers to LDS buffer `cur` ----
        *(uint4*)&sA[cur][arow * 32 + acol]     = ra0;
        *(uint4*)&sA[cur][arow * 32 + acol + 8] = ra1;
        #pragma unroll
        for (int i = 0; i < 4; ++i) {
            int kk = 2 * wp + 8 * i;
            unsigned int pk = (unsigned int)f2bf(wf[2 * i])
                            | ((unsigned int)f2bf(wf[2 * i + 1]) << 16);
            *(unsigned int*)&sW[cur][nn * 32 + kk] = pk;
        }
        __syncthreads();

        // ---- stage NEXT K-step into registers (overlaps with WMMAs) ----
        if (ks + 1 < nk) {
            int k0 = (ks + 1) * 32;
            const unsigned short* bp = (k0 < ksplit ? aptrA : aptrB) + k0;
            ra0 = *(const uint4*)(bp + acol);
            ra1 = *(const uint4*)(bp + acol + 8);
            const float* wsel;
            size_t kro;
            if (k0 < ksplit) { wsel = W0; kro = (size_t)k0; }
            else             { wsel = W1; kro = (size_t)(k0 - ksplit); }
            const float* wbase = wsel + kro * ldw + bx * 64 + nn;
            #pragma unroll
            for (int i = 0; i < 4; ++i) {
                int kk = 2 * wp + 8 * i;
                wf[2 * i]     = wbase[(size_t)kk * ldw];
                wf[2 * i + 1] = wbase[(size_t)(kk + 1) * ldw];
            }
            if (ks + 2 < nk) {
                int k2 = (ks + 2) * 32;
                __builtin_prefetch((k2 < ksplit ? aptrA : aptrB) + k2 + acol, 0, 1);
            }
        }

        // ---- compute: load A frag (2 x b128) + 4 B frags (8 x b128),
        //      then 4 back-to-back WMMAs (single dscnt drain) ----
        const int m  = lane & 15;
        const int g  = lane >> 4;
        union { bf16x16 v; uint4 q[2]; } fa;
        {
            const unsigned short* base = &sA[cur][(wm * 16 + m) * 32 + 8 * g];
            fa.q[0] = *(const uint4*)&base[0];     // K halves 0..7  (+8g)
            fa.q[1] = *(const uint4*)&base[16];    // K halves 16..23 (+8g)
        }
        union { bf16x16 v; uint4 q[2]; } fb[4];
        #pragma unroll
        for (int t = 0; t < 4; ++t) {
            const unsigned short* p = &sW[cur][(t * 16 + m) * 32 + g * 8];
            fb[t].q[0] = *(const uint4*)&p[0];
            fb[t].q[1] = *(const uint4*)&p[16];
        }
        #pragma unroll
        for (int t = 0; t < 4; ++t) {
            acc[t] = __builtin_amdgcn_wmma_f32_16x16x32_bf16(false, fa.v, false, fb[t].v,
                                                             (short)0, acc[t], false, false);
        }
        cur ^= 1;
    }

    // ---- epilogue: bias, gather-add, activation, stores ----
    const int g  = lane >> 4;
    const int nl = lane & 15;
    #pragma unroll
    for (int r = 0; r < 8; ++r) {
        int mrow = by * 128 + wm * 16 + g * 8 + r;
        const float* grow = nullptr;
        if (G) grow = G + (size_t)(gidx ? gidx[mrow] : mrow) * ldg;
        #pragma unroll
        for (int t = 0; t < 4; ++t) {
            int ncol = bx * 64 + t * 16 + nl;
            float v = acc[t][r] + (b0 ? b0[ncol] : 0.0f) + (b1 ? b1[ncol] : 0.0f);
            if (grow) v += grow[ncol];
            if (act == 1)      v = (v > 0.0f) ? v : 0.2f * v;
            else if (act == 2) v = gelu_f(v);
            if (C32) C32[(size_t)mrow * ldc + ncol] = v;
            if (Cb)  Cb[(size_t)mrow * ldc + ncol]  = f2bf(v);
        }
    }
}

// ---------------------------------------------------------------------------
// fp32 -> bf16 buffer conversion
// ---------------------------------------------------------------------------
__global__ void k_cvt(const float* __restrict__ in, unsigned short* __restrict__ out, int n) {
    int i = blockIdx.x * blockDim.x + threadIdx.x;
    if (i < n) out[i] = f2bf(in[i]);
}

// ---------------------------------------------------------------------------
// init: amax=-inf, den=0, agg=0  (must run each call: graph-safe determinism)
// ---------------------------------------------------------------------------
__global__ void k_init(float* amax, float* den, float* agg, int nh, int nd) {
    int i = blockIdx.x * blockDim.x + threadIdx.x;
    if (i < nh) { amax[i] = -__builtin_inff(); den[i] = 0.0f; }
    if (i < nd) agg[i] = 0.0f;
}

// ---------------------------------------------------------------------------
// Per-node stats: ||mem_hidden|| (clamped), sqrt(mean(mem_var)),
// evolution-bank temporal consistency.  One wave32 per node.
// ---------------------------------------------------------------------------
__global__ __launch_bounds__(256) void k_node_stats(
    const float* __restrict__ memh, const float* __restrict__ memv,
    const float* __restrict__ evo,
    float* __restrict__ hnorm, float* __restrict__ unc, float* __restrict__ tc, int nN)
{
    int gw   = (blockIdx.x * 256 + threadIdx.x) >> 5;
    int lane = threadIdx.x & 31;
    if (gw >= nN) return;

    const float* hrow = memh + (size_t)gw * 256;
    const float* vrow = memv + (size_t)gw * 256;
    float s = 0.f, sv = 0.f;
    #pragma unroll
    for (int i = 0; i < 8; ++i) {
        float h = hrow[lane + i * 32];
        s  += h * h;
        sv += vrow[lane + i * 32];
    }
    #pragma unroll
    for (int off = 16; off > 0; off >>= 1) {
        s  += __shfl_xor(s,  off, 32);
        sv += __shfl_xor(sv, off, 32);
    }
    if (lane == 0) {
        hnorm[gw] = fmaxf(sqrtf(s), 1e-6f);
        unc[gw]   = sqrtf(sv * (1.0f / 256.0f));
    }

    // temporal consistency over W=6 bank rows
    float prev[8];
    float prevn = 1.f;
    float sims[5];
    #pragma unroll
    for (int w = 0; w < 6; ++w) {
        const float* er = evo + ((size_t)gw * 6 + w) * 256;
        float cur[8]; float ss = 0.f, dt = 0.f;
        #pragma unroll
        for (int i = 0; i < 8; ++i) { cur[i] = er[lane + i * 32]; ss += cur[i] * cur[i]; }
        if (w > 0) {
            #pragma unroll
            for (int i = 0; i < 8; ++i) dt += prev[i] * cur[i];
        }
        #pragma unroll
        for (int off = 16; off > 0; off >>= 1) {
            ss += __shfl_xor(ss, off, 32);
            dt += __shfl_xor(dt, off, 32);
        }
        float nrm = fmaxf(sqrtf(ss), 1e-6f);
        if (w > 0) sims[w - 1] = dt / (prevn * nrm);
        prevn = nrm;
        #pragma unroll
        for (int i = 0; i < 8; ++i) prev[i] = cur[i];
    }
    float mean = 0.f;
    #pragma unroll
    for (int i = 0; i < 5; ++i) mean += sims[i];
    mean *= 0.2f;
    float var = 0.f;
    #pragma unroll
    for (int i = 0; i < 5; ++i) { float d = sims[i] - mean; var += d * d; }
    var *= 0.25f;                                    // ddof=1 -> /4
    float t = 1.0f / (1.0f + sqrtf(var));
    if (lane == 0) tc[gw] = fminf(fmaxf(t, 0.0f), 1.0f);
}

// ---------------------------------------------------------------------------
// Row-normalize (rows of length 64), l2n with eps clamp.  One wave per row.
// ---------------------------------------------------------------------------
__global__ __launch_bounds__(256) void k_rownorm64(float* __restrict__ buf, int rows) {
    int gw   = (blockIdx.x * 256 + threadIdx.x) >> 5;
    int lane = threadIdx.x & 31;
    if (gw >= rows) return;
    float a = buf[(size_t)gw * 64 + lane];
    float b = buf[(size_t)gw * 64 + lane + 32];
    float s = a * a + b * b;
    #pragma unroll
    for (int off = 16; off > 0; off >>= 1) s += __shfl_xor(s, off, 32);
    float inv = 1.0f / fmaxf(sqrtf(s), 1e-6f);
    buf[(size_t)gw * 64 + lane]      = a * inv;
    buf[(size_t)gw * 64 + lane + 32] = b * inv;
}

// ---------------------------------------------------------------------------
// Per-edge: attention logits (h1 @ Wa2 + ba2) and structural trust MLP,
// writes a[e,h] = logit * trust.  One wave32 per edge, 8 waves / block.
// Wm2 (64x64) is staged once into LDS and shared by the block's 8 edges.
// ---------------------------------------------------------------------------
__global__ __launch_bounds__(256) void k_attn_trust(
    const unsigned short* __restrict__ h1b,        // E x 256 bf16
    const float* __restrict__ Wa2, const float* __restrict__ ba2,
    const float* __restrict__ memh, const float* __restrict__ hnorm,
    const float* __restrict__ ni, const float* __restrict__ nj,
    const float* __restrict__ unc, const float* __restrict__ tc,
    const int* __restrict__ deg,
    const int* __restrict__ srcI, const int* __restrict__ dstI,
    const float* __restrict__ Wm1, const float* __restrict__ bm1,
    const float* __restrict__ Wm2, const float* __restrict__ bm2,
    const float* __restrict__ Wm3, const float* __restrict__ bm3,
    const float* __restrict__ log_temp,
    float* __restrict__ aout)
{
    __shared__ float sW2[64 * 64];                 // 16 KB, shared by 8 edges
    __shared__ float sg[8][64];
    int wv   = threadIdx.x >> 5;
    int lane = threadIdx.x & 31;
    int e = blockIdx.x * 8 + wv;
    int s = srcI[e], d = dstI[e];

    for (int i = threadIdx.x; i < 64 * 64; i += 256) sW2[i] = Wm2[i];

    // attention logits: 8 heads, dot over 256 with Wa2 (256x8)
    float ph[8];
    #pragma unroll
    for (int h = 0; h < 8; ++h) ph[h] = 0.f;
    {
        union { uint4 q; unsigned short us[8]; } ld;
        ld.q = *(const uint4*)&h1b[(size_t)e * 256 + lane * 8];
        #pragma unroll
        for (int i = 0; i < 8; ++i) {
            float xv = bf2f(ld.us[i]);
            const float* wr = &Wa2[(lane * 8 + i) * 8];
            #pragma unroll
            for (int h = 0; h < 8; ++h) ph[h] += xv * wr[h];
        }
    }
    #pragma unroll
    for (int off = 16; off > 0; off >>= 1) {
        #pragma unroll
        for (int h = 0; h < 8; ++h) ph[h] += __shfl_xor(ph[h], off, 32);
    }

    // trust features
    float dt = 0.f;
    const float* hd = memh + (size_t)d * 256;
    const float* hs = memh + (size_t)s * 256;
    #pragma unroll
    for (int i = 0; i < 8; ++i) dt += hd[lane + i * 32] * hs[lane + i * 32];
    float ncs = ni[(size_t)d * 64 + lane] * nj[(size_t)s * 64 + lane]
              + ni[(size_t)d * 64 + lane + 32] * nj[(size_t)s * 64 + lane + 32];
    #pragma unroll
    for (int off = 16; off > 0; off >>= 1) {
        dt  += __shfl_xor(dt,  off, 32);
        ncs += __shfl_xor(ncs, off, 32);
    }
    float ec = dt / (hnorm[d] * hnorm[s]);
    float di = (float)deg[d], dj = (float)deg[s];
    float dsim = 1.0f - fabsf(di - dj) / (di + dj + 1e-6f);
    float f0 = ec, f1 = dsim, f2 = ncs, f3 = tc[s], f4 = unc[d], f5 = unc[s];

    // trust MLP 6->64 (gelu), each lane owns units lane and lane+32
    #pragma unroll
    for (int t = 0; t < 2; ++t) {
        int u = lane + t * 32;
        float acc = bm1[u]
            + f0 * Wm1[0 * 64 + u] + f1 * Wm1[1 * 64 + u] + f2 * Wm1[2 * 64 + u]
            + f3 * Wm1[3 * 64 + u] + f4 * Wm1[4 * 64 + u] + f5 * Wm1[5 * 64 + u];
        sg[wv][u] = gelu_f(acc);
    }
    __syncthreads();
    // 64->64 (gelu) then 64->1
    float part = 0.f;
    #pragma unroll
    for (int t = 0; t < 2; ++t) {
        int u = lane + t * 32;
        float acc = bm2[u];
        for (int k = 0; k < 64; ++k) acc += sg[wv][k] * sW2[k * 64 + u];
        part += gelu_f(acc) * Wm3[u];
    }
    #pragma unroll
    for (int off = 16; off > 0; off >>= 1) part += __shfl_xor(part, off, 32);
    float logit = part + bm3[0];
    float temp  = fminf(fmaxf(expf(log_temp[0]), 0.1f), 10.0f);
    float trust = 1.0f / (1.0f + expf(-logit * temp));

    float av = ph[0];
    #pragma unroll
    for (int h = 1; h < 8; ++h) if (lane == h) av = ph[h];
    if (lane < 8) aout[(size_t)e * 8 + lane] = (av + ba2[lane]) * trust;
}

// ---------------------------------------------------------------------------
// segment softmax (per dst node, per head) + aggregation
// ---------------------------------------------------------------------------
__global__ void k_segmax(const float* __restrict__ a, const int* __restrict__ dstI,
                         float* __restrict__ amax, int EH) {
    int i = blockIdx.x * blockDim.x + threadIdx.x;
    if (i >= EH) return;
    int e = i >> 3, h = i & 7;
    atomicMaxF(&amax[dstI[e] * 8 + h], a[i]);
}

__global__ void k_expden(float* __restrict__ a, const int* __restrict__ dstI,
                         const float* __restrict__ amax, float* __restrict__ den, int EH) {
    int i = blockIdx.x * blockDim.x + threadIdx.x;
    if (i >= EH) return;
    int e = i >> 3, h = i & 7;
    float ex = expf(a[i] - amax[dstI[e] * 8 + h]);
    a[i] = ex;
    atomicAdd(&den[dstI[e] * 8 + h], ex);
}

__global__ __launch_bounds__(256) void k_agg(
    const float* __restrict__ a, const float* __restrict__ den,
    const float* __restrict__ V, const int* __restrict__ srcI,
    const int* __restrict__ dstI, float* __restrict__ agg)
{
    int e = blockIdx.x, c = threadIdx.x, h = c >> 5;   // Dh = 32
    int s = srcI[e], d = dstI[e];
    float at = a[(size_t)e * 8 + h] / (den[d * 8 + h] + 1e-16f);
    atomicAdd(&agg[(size_t)d * 256 + c], V[(size_t)s * 256 + c] * at);
}

// ---------------------------------------------------------------------------
// gate + residual + layernorm; one 256-thread block per node
// ---------------------------------------------------------------------------
__global__ __launch_bounds__(256) void k_final(
    const float* __restrict__ x, const float* __restrict__ o,
    const float* __restrict__ Wg, const float* __restrict__ bg,
    const float* __restrict__ gamma, const float* __restrict__ beta,
    float* __restrict__ y)
{
    int n = blockIdx.x, c = threadIdx.x;
    __shared__ float red[256];
    float xv = x[(size_t)n * 256 + c];
    float ov = o[(size_t)n * 256 + c];

    red[c] = xv * Wg[c];
    __syncthreads();
    for (int s2 = 128; s2 > 0; s2 >>= 1) { if (c < s2) red[c] += red[c + s2]; __syncthreads(); }
    float g = 1.0f / (1.0f + expf(-(red[0] + bg[0])));
    float yv = g * xv + (1.0f - g) * ov;
    __syncthreads();

    red[c] = yv;
    __syncthreads();
    for (int s2 = 128; s2 > 0; s2 >>= 1) { if (c < s2) red[c] += red[c + s2]; __syncthreads(); }
    float mu = red[0] * (1.0f / 256.0f);
    __syncthreads();

    float dv = yv - mu;
    red[c] = dv * dv;
    __syncthreads();
    for (int s2 = 128; s2 > 0; s2 >>= 1) { if (c < s2) red[c] += red[c + s2]; __syncthreads(); }
    float var = red[0] * (1.0f / 256.0f);
    y[(size_t)n * 256 + c] = dv * rsqrtf(var + 1e-5f) * gamma[c] + beta[c];
}

// ---------------------------------------------------------------------------
// Host orchestration
// ---------------------------------------------------------------------------
extern "C" void kernel_launch(void* const* d_in, const int* in_sizes, int n_in,
                              void* d_out, int out_size, void* d_ws, size_t ws_size,
                              hipStream_t stream) {
    (void)in_sizes; (void)n_in; (void)out_size; (void)ws_size;
    const int N = NN_, E = EE_, D = DD_, H = HH_, HD = HD_;

    const float* x        = (const float*)d_in[0];
    const int*   eidx     = (const int*)d_in[1];
    const float* edge_emb = (const float*)d_in[2];
    const float* time_emb = (const float*)d_in[3];
    const int*   deg      = (const int*)d_in[4];
    const float* memh     = (const float*)d_in[5];
    const float* memv     = (const float*)d_in[6];
    const float* evo      = (const float*)d_in[7];
    const float* Wq = (const float*)d_in[8];   const float* bq = (const float*)d_in[9];
    const float* Wk = (const float*)d_in[10];  const float* bk = (const float*)d_in[11];
    const float* Wv = (const float*)d_in[12];  const float* bv = (const float*)d_in[13];
    const float* Wo = (const float*)d_in[14];  const float* bo = (const float*)d_in[15];
    const float* We = (const float*)d_in[16];  const float* be = (const float*)d_in[17];
    const float* Wt = (const float*)d_in[18];  const float* bt = (const float*)d_in[19];
    const float* Wa1 = (const float*)d_in[20]; const float* ba1 = (const float*)d_in[21];
    const float* Wa2 = (const float*)d_in[22]; const float* ba2 = (const float*)d_in[23];
    const float* Wni = (const float*)d_in[24]; const float* bni = (const float*)d_in[25];
    const float* Wnj = (const float*)d_in[26]; const float* bnj = (const float*)d_in[27];
    const float* Wm1 = (const float*)d_in[28]; const float* bm1 = (const float*)d_in[29];
    const float* Wm2 = (const float*)d_in[30]; const float* bm2 = (const float*)d_in[31];
    const float* Wm3 = (const float*)d_in[32]; const float* bm3 = (const float*)d_in[33];
    const float* log_temp = (const float*)d_in[34];
    const float* Wg = (const float*)d_in[35];  const float* bg = (const float*)d_in[36];
    const float* Wf1 = (const float*)d_in[37]; const float* bf1 = (const float*)d_in[38];
    const float* Wf2 = (const float*)d_in[39]; const float* bf2 = (const float*)d_in[40];
    const float* gamma = (const float*)d_in[41]; const float* beta = (const float*)d_in[42];

    const int* srcI = eidx;
    const int* dstI = eidx + E;

    // ---- carve workspace (aliased where lifetimes allow) ----
    char* wp = (char*)d_ws;
    auto alloc = [&](size_t bytes) -> char* {
        char* p = wp;
        wp += (bytes + 255) & ~(size_t)255;
        return p;
    };
    unsigned short* xb    = (unsigned short*)alloc((size_t)N * D * 2);
    unsigned short* mb    = (unsigned short*)alloc((size_t)N * D * 2);
    unsigned short* eb    = (unsigned short*)alloc((size_t)E * D * 2);  // -> h1b later
    unsigned short* tb    = (unsigned short*)alloc((size_t)E * D * 2);  // -> f1b/f2b/o32 later
    unsigned short* Qb    = (unsigned short*)alloc((size_t)N * D * 2);  // -> outb later
    float* K32  = (float*)alloc((size_t)N * D * 4);
    float* V32  = (float*)alloc((size_t)N * D * 4);
    unsigned short* krich = (unsigned short*)alloc((size_t)E * D * 2);
    float* aE   = (float*)alloc((size_t)E * H * 4);
    float* ni32 = (float*)alloc((size_t)N * HD * 4);
    float* nj32 = (float*)alloc((size_t)N * HD * 4);
    float* hnorm = (float*)alloc((size_t)N * 4);
    float* unc   = (float*)alloc((size_t)N * 4);
    float* tcb   = (float*)alloc((size_t)N * 4);
    float* amax  = (float*)alloc((size_t)N * H * 4);
    float* den   = (float*)alloc((size_t)N * H * 4);
    float* agg   = (float*)alloc((size_t)N * D * 4);
    // lifetime-safe aliases
    unsigned short* h1b  = eb;                            // eb dead after eete gemm
    unsigned short* f1b  = tb;                            // tb dead after eete gemm
    unsigned short* f2b  = tb + (size_t)N * D;
    float*          o32  = (float*)(tb + (size_t)2 * N * D);
    unsigned short* outb = Qb;                            // Qb dead after attn1 gemm

    auto gemm = [&](const unsigned short* A0_, int lda0_,
                    const unsigned short* A1_, int lda1_,
                    const int* i0, const int* i1, int ksp, int Ktot_,
                    const float* W0_, const float* W1_, int ldw_,
                    const float* b0_, const float* b1_,
                    const float* G_, const int* gi, int ldg_,
                    int M_, int Ncols_, int act_,
                    unsigned short* Cb_, float* C32_, int ldc_) {
        dim3 grid(Ncols_ / 64, M_ / 128);
        k_gemm<<<grid, 256, 0, stream>>>(A0_, lda0_, A1_, lda1_, i0, i1, ksp, Ktot_,
                                         W0_, W1_, ldw_, b0_, b1_, G_, gi, ldg_,
                                         act_, Cb_, C32_, ldc_);
    };

    // 0) re-init accumulators each call (deterministic, graph-safe)
    k_init<<<dim3((N * D + 255) / 256), 256, 0, stream>>>(amax, den, agg, N * H, N * D);

    // 1) fp32 -> bf16 staging
    k_cvt<<<dim3((N * D + 255) / 256), 256, 0, stream>>>(x, xb, N * D);
    k_cvt<<<dim3((N * D + 255) / 256), 256, 0, stream>>>(memh, mb, N * D);
    k_cvt<<<dim3((E * D + 255) / 256), 256, 0, stream>>>(edge_emb, eb, E * D);
    k_cvt<<<dim3((E * D + 255) / 256), 256, 0, stream>>>(time_emb, tb, E * D);

    // 2) per-node stats
    k_node_stats<<<dim3(N / 8), 256, 0, stream>>>(memh, memv, evo, hnorm, unc, tcb, N);

    // 3) node GEMMs: Q (bf16), K (fp32 for gather-add), V (fp32)
    gemm(xb,256, nullptr,0, nullptr,nullptr, 256,256, Wq,nullptr,256, bq,nullptr,
         nullptr,nullptr,0, N,256,0, Qb,nullptr,256);
    gemm(xb,256, nullptr,0, nullptr,nullptr, 256,256, Wk,nullptr,256, bk,nullptr,
         nullptr,nullptr,0, N,256,0, nullptr,K32,256);
    gemm(xb,256, nullptr,0, nullptr,nullptr, 256,256, Wv,nullptr,256, bv,nullptr,
         nullptr,nullptr,0, N,256,0, nullptr,V32,256);

    // 4) trust projections ni/nj + l2 row-normalize
    gemm(mb,256, nullptr,0, nullptr,nullptr, 256,256, Wni,nullptr,64, bni,nullptr,
         nullptr,nullptr,0, N,64,0, nullptr,ni32,64);
    k_rownorm64<<<dim3(N / 8), 256, 0, stream>>>(ni32, N);
    gemm(mb,256, nullptr,0, nullptr,nullptr, 256,256, Wnj,nullptr,64, bnj,nullptr,
         nullptr,nullptr,0, N,64,0, nullptr,nj32,64);
    k_rownorm64<<<dim3(N / 8), 256, 0, stream>>>(nj32, N);

    // 5) K_rich = (edge_emb@We+be) + (time_emb@Wt+bt) + K[src]   (K-concat GEMM)
    gemm(eb,256, tb,256, nullptr,nullptr, 256,512, We,Wt,256, be,bt,
         K32,srcI,256, E,256,0, krich,nullptr,256);

    // 6) h1 = leaky_relu([Q[dst] | K_rich] @ Wa1 + ba1)
    gemm(Qb,256, krich,256, dstI,nullptr, 256,512, Wa1,Wa1 + 256 * 256,256, ba1,nullptr,
         nullptr,nullptr,0, E,256,1, h1b,nullptr,256);

    // 7) attention head logits + structural trust, a *= trust
    k_attn_trust<<<dim3(E / 8), 256, 0, stream>>>(h1b, Wa2, ba2, memh, hnorm, ni32, nj32,
                                                  unc, tcb, deg, srcI, dstI,
                                                  Wm1, bm1, Wm2, bm2, Wm3, bm3,
                                                  log_temp, aE);

    // 8) segment softmax over dst per head + message aggregation
    k_segmax<<<dim3((E * H + 255) / 256), 256, 0, stream>>>(aE, dstI, amax, E * H);
    k_expden<<<dim3((E * H + 255) / 256), 256, 0, stream>>>(aE, dstI, amax, den, E * H);
    k_agg<<<dim3(E), 256, 0, stream>>>(aE, den, V32, srcI, dstI, agg);

    // 9) fusion chain: gelu([out|mem_hidden]@Wf1+bf1)@Wf2+bf2 then @Wo+bo
    k_cvt<<<dim3((N * D + 255) / 256), 256, 0, stream>>>(agg, outb, N * D);
    gemm(outb,256, mb,256, nullptr,nullptr, 256,512, Wf1,Wf1 + 256 * 256,256, bf1,nullptr,
         nullptr,nullptr,0, N,256,2, f1b,nullptr,256);
    gemm(f1b,256, nullptr,0, nullptr,nullptr, 256,256, Wf2,nullptr,256, bf2,nullptr,
         nullptr,nullptr,0, N,256,0, f2b,nullptr,256);
    gemm(f2b,256, nullptr,0, nullptr,nullptr, 256,256, Wo,nullptr,256, bo,nullptr,
         nullptr,nullptr,0, N,256,0, nullptr,o32,256);

    // 10) gate + residual + layernorm -> output
    k_final<<<dim3(N), 256, 0, stream>>>(x, o32, Wg, bg, gamma, beta, (float*)d_out);
}
